// Model_8710193676408
// MI455X (gfx1250) — matrix operand order repaired and checked by reading, not compiled, not run on hardware
//
#include <hip/hip_runtime.h>

#define N_NODES 20000
#define N_EDGES 500000
#define N_FEAT  2048
#define N_HID   128
#define N_CLASS 20

// K-chunk staged in LDS per workgroup (bf16): 128 cols x 128 k = 32 KB.
#define KC   128
// Padded LDS row stride: +16B breaks the 256B-stride bank aliasing
// (lane m16 -> bank 4*m16) while preserving 16B alignment for ds_load_b128.
#define KCP  (KC + 8)

typedef __attribute__((ext_vector_type(16))) __bf16 v16bf;
typedef __attribute__((ext_vector_type(8)))  float  v8f;
typedef __attribute__((ext_vector_type(4)))  float  v4f;

// ---------------------------------------------------------------------------
// Convert fp32 weight [K,N] (row-major) -> bf16 transposed [N,K] so that each
// WMMA B-fragment (col n, 16 consecutive K) is one contiguous 32B run.
// ---------------------------------------------------------------------------
__global__ void cvt_transpose_kernel(const float* __restrict__ W,
                                     __bf16* __restrict__ Wt, int K, int N) {
    int idx = blockIdx.x * blockDim.x + threadIdx.x;
    if (idx >= K * N) return;
    int k = idx / N, n = idx % N;
    Wt[(size_t)n * K + k] = (__bf16)W[(size_t)k * N + n];
}

// ---------------------------------------------------------------------------
// CSR row_ptr from sorted edge-row array via per-row lower_bound.
// ---------------------------------------------------------------------------
__global__ void rowptr_kernel(const int* __restrict__ rows, int* __restrict__ rp) {
    int r = blockIdx.x * blockDim.x + threadIdx.x;
    if (r > N_NODES) return;
    int lo = 0, hi = N_EDGES;
    while (lo < hi) {
        int mid = (lo + hi) >> 1;
        if (rows[mid] < r) lo = mid + 1; else hi = mid;
    }
    rp[r] = lo;
}

// ---------------------------------------------------------------------------
// C[M,128] = A[M,K] (fp32 -> bf16 in-register) @ Bt (bf16 [128,K] transposed).
// 256 threads = 8 waves; each wave computes a 16-row strip x 128 cols with
// 8x v_wmma_f32_16x16x32_bf16 per K-step.
//   - B staged through double-buffered LDS (32KB/chunk): the 8 waves of a WG
//     share one L2 fetch of each weight chunk.
//   - Register-level pipelining: B fragment nt+1 is ds_loaded during WMMA nt;
//     A fragment for K-step k+32 is global-loaded before the WMMA burst of k.
//
// Fragment layouts used here (wave32; contiguous k order, same for A and B):
//   A 16x32 bf16 : lane l -> row (l&15); half=(l>>4): K =
//                  k0 + half*16 + [0..15] (one 64B fp32 run)
//   B 32x16 bf16 : lane l -> col (l&15); K = k0 + (l>>4)*16 + [0..15]
//   C/D f32 16x16: vgpr r, lane l -> (M=(l>>4)*8+r, N=l&15)
// ---------------------------------------------------------------------------
__global__ __launch_bounds__(256)
void gemm_bf16_wmma_kernel(const float* __restrict__ A,
                           const __bf16* __restrict__ Bt,
                           float* __restrict__ C, int M, int K) {
    __shared__ __bf16 Bs[2][128 * KCP];   // 2 x 34 KB

    const int lane = threadIdx.x & 31;
    const int wave = threadIdx.x >> 5;
    const int m16  = lane & 15;
    const int half = lane >> 4;
    const int rowBase = blockIdx.x * 128 + wave * 16;

    v8f acc[8];
#pragma unroll
    for (int i = 0; i < 8; ++i) acc[i] = (v8f){0.f,0.f,0.f,0.f,0.f,0.f,0.f,0.f};

    const int arow = min(rowBase + m16, M - 1);
    const float* __restrict__ ap = A + (size_t)arow * K + half * 16;

    const int numChunks = K / KC;

    // ---- preload LDS chunk 0 (32KB: 2048 x 16B, 8 per thread) ----
#pragma unroll
    for (int j = 0; j < 8; ++j) {
        int i  = threadIdx.x + j * 256;
        int n  = i >> 4;
        int ko = (i & 15) * 8;                       // bf16 offset within row
        uint4 v = *(const uint4*)(Bt + (size_t)n * K + ko);
        *(uint4*)(&Bs[0][n * KCP + ko]) = v;
    }

    // ---- preload A registers for the very first K-step ----
    v4f a0 = *(const v4f*)(ap);
    v4f a1 = *(const v4f*)(ap + 4);
    v4f a2 = *(const v4f*)(ap + 8);
    v4f a3 = *(const v4f*)(ap + 12);

    __syncthreads();

    for (int c = 0; c < numChunks; ++c) {
        const int buf = c & 1;
        const bool hasNext = (c + 1 < numChunks);

        // ---- issue global loads for LDS chunk c+1 (in flight over compute)
        uint4 tmp[8];
        if (hasNext) {
            const int k0n = (c + 1) * KC;
#pragma unroll
            for (int j = 0; j < 8; ++j) {
                int i  = threadIdx.x + j * 256;
                int n  = i >> 4;
                int ko = (i & 15) * 8;
                tmp[j] = *(const uint4*)(Bt + (size_t)n * K + k0n + ko);
            }
        }

        // ---- compute 4 K-steps of 32 out of Bs[buf] ----
        const int k0 = c * KC;
        const __bf16* __restrict__ bs = &Bs[buf][0];
#pragma unroll
        for (int ks = 0; ks < KC; ks += 32) {
            // cvt previously-loaded A chunk -> bf16 fragment
            v16bf af;
            af[0]  = (__bf16)a0[0]; af[1]  = (__bf16)a0[1];
            af[2]  = (__bf16)a0[2]; af[3]  = (__bf16)a0[3];
            af[4]  = (__bf16)a1[0]; af[5]  = (__bf16)a1[1];
            af[6]  = (__bf16)a1[2]; af[7]  = (__bf16)a1[3];
            af[8]  = (__bf16)a2[0]; af[9]  = (__bf16)a2[1];
            af[10] = (__bf16)a2[2]; af[11] = (__bf16)a2[3];
            af[12] = (__bf16)a3[0]; af[13] = (__bf16)a3[1];
            af[14] = (__bf16)a3[2]; af[15] = (__bf16)a3[3];

            // prefetch A for next K-step (clamped; lands during WMMA burst)
            {
                int knext = k0 + ks + 32;
                if (knext > K - 32) knext = K - 32;
                const float* apn = ap + knext;
                a0 = *(const v4f*)(apn);
                a1 = *(const v4f*)(apn + 4);
                a2 = *(const v4f*)(apn + 8);
                a3 = *(const v4f*)(apn + 12);
            }

            const int kb = ks + half * 16;
            // register-level double buffer of B fragments
            v16bf bcur = *(const v16bf*)(bs + m16 * KCP + kb);
#pragma unroll
            for (int nt = 0; nt < 8; ++nt) {
                v16bf bnxt = bcur;
                if (nt < 7)
                    bnxt = *(const v16bf*)(bs + ((nt + 1) * 16 + m16) * KCP + kb);
                acc[nt] = __builtin_amdgcn_wmma_f32_16x16x32_bf16(
                    false, af, false, bcur, (short)0, acc[nt], false, false);
                bcur = bnxt;
            }
        }

        // ---- commit prefetched chunk c+1 to the other LDS buffer ----
        if (hasNext) {
#pragma unroll
            for (int j = 0; j < 8; ++j) {
                int i  = threadIdx.x + j * 256;
                int n  = i >> 4;
                int ko = (i & 15) * 8;
                *(uint4*)(&Bs[buf ^ 1][n * KCP + ko]) = tmp[j];
            }
        }
        __syncthreads();
    }

    // ---- store D (f32 16x16 layout) ----
#pragma unroll
    for (int nt = 0; nt < 8; ++nt) {
#pragma unroll
        for (int r = 0; r < 8; ++r) {
            int row = rowBase + half * 8 + r;
            if (row < M) C[(size_t)row * N_HID + nt * 16 + m16] = acc[nt][r];
        }
    }
}

// ---------------------------------------------------------------------------
// out[r,:] = sum_{e in row r} val[e] * dense[col[e],:] + bias  (opt. relu)
// One block per row, thread-per-feature: 512B contiguous gather per edge,
// working set (10.2 MB) lives in L2. Deterministic (no atomics). Edge loop
// unrolled x2 with dual accumulators to shorten the gather->FMA chain.
// ---------------------------------------------------------------------------
__global__ __launch_bounds__(128)
void spmm_kernel(const int* __restrict__ rp, const int* __restrict__ col,
                 const float* __restrict__ val, const float* __restrict__ dense,
                 const float* __restrict__ bias, float* __restrict__ out,
                 int relu) {
    const int r = blockIdx.x;
    const int t = threadIdx.x;
    const int e0 = rp[r], e1 = rp[r + 1];
    float acc0 = 0.f, acc1 = 0.f;
    int e = e0;
    for (; e + 1 < e1; e += 2) {
        acc0 = fmaf(val[e],     dense[(size_t)col[e]     * N_HID + t], acc0);
        acc1 = fmaf(val[e + 1], dense[(size_t)col[e + 1] * N_HID + t], acc1);
    }
    if (e < e1)
        acc0 = fmaf(val[e], dense[(size_t)col[e] * N_HID + t], acc0);
    float acc = acc0 + acc1 + bias[t];
    if (relu) acc = fmaxf(acc, 0.f);
    out[(size_t)r * N_HID + t] = acc;
}

// ---------------------------------------------------------------------------
// y[n,c] = h[n,:] . fc_w[:,c] + fc_b[c]   (N_CLASS=20: plain VALU dot)
// ---------------------------------------------------------------------------
__global__ void fc_kernel(const float* __restrict__ h, const float* __restrict__ W,
                          const float* __restrict__ b, float* __restrict__ y) {
    int idx = blockIdx.x * blockDim.x + threadIdx.x;
    if (idx >= N_NODES * N_CLASS) return;
    int n = idx / N_CLASS, c = idx % N_CLASS;
    const float* __restrict__ hr = h + (size_t)n * N_HID;
    float acc = b[c];
#pragma unroll 8
    for (int k = 0; k < N_HID; ++k) acc = fmaf(hr[k], W[k * N_CLASS + c], acc);
    y[idx] = acc;
}

// ---------------------------------------------------------------------------
extern "C" void kernel_launch(void* const* d_in, const int* in_sizes, int n_in,
                              void* d_out, int out_size, void* d_ws, size_t ws_size,
                              hipStream_t stream) {
    const float* x    = (const float*)d_in[0];
    const int*   arow = (const int*)  d_in[1];
    const int*   acol = (const int*)  d_in[2];
    const float* aval = (const float*)d_in[3];
    const float* W1   = (const float*)d_in[4];
    const float* b1   = (const float*)d_in[5];
    const float* W2   = (const float*)d_in[6];
    const float* b2   = (const float*)d_in[7];
    const float* fcw  = (const float*)d_in[8];
    const float* fcb  = (const float*)d_in[9];

    float* out_h = (float*)d_out;                      // [20000,128]
    float* out_y = out_h + (size_t)N_NODES * N_HID;    // [20000,20]

    char* ws = (char*)d_ws;
    __bf16* W1t = (__bf16*)ws;  ws += (((size_t)N_HID * N_FEAT * 2) + 255) & ~(size_t)255;
    __bf16* W2t = (__bf16*)ws;  ws += (((size_t)N_HID * N_HID * 2) + 255) & ~(size_t)255;
    int*    rp  = (int*)ws;     ws += (((size_t)(N_NODES + 1) * 4) + 255) & ~(size_t)255;
    float*  sup1 = (float*)ws;  ws += (size_t)N_NODES * N_HID * 4;
    float*  h1   = (float*)ws;  ws += (size_t)N_NODES * N_HID * 4;
    float*  sup2 = (float*)ws;

    // Pre-convert weights to bf16 transposed (W1: 1MB fp32 -> 512KB bf16).
    cvt_transpose_kernel<<<(N_FEAT * N_HID + 255) / 256, 256, 0, stream>>>(W1, W1t, N_FEAT, N_HID);
    cvt_transpose_kernel<<<(N_HID * N_HID + 255) / 256, 256, 0, stream>>>(W2, W2t, N_HID, N_HID);
    rowptr_kernel<<<(N_NODES + 1 + 255) / 256, 256, 0, stream>>>(arow, rp);

    const int gemm_blocks = (N_NODES + 127) / 128;

    // Layer 1: support1 = x @ W1 (WMMA), h1 = spmm + b1, relu
    gemm_bf16_wmma_kernel<<<gemm_blocks, 256, 0, stream>>>(x, W1t, sup1, N_NODES, N_FEAT);
    spmm_kernel<<<N_NODES, 128, 0, stream>>>(rp, acol, aval, sup1, b1, h1, 1);

    // Layer 2: support2 = h1 @ W2 (WMMA), h2 = spmm + b2 -> out_h
    gemm_bf16_wmma_kernel<<<gemm_blocks, 256, 0, stream>>>(h1, W2t, sup2, N_NODES, N_HID);
    spmm_kernel<<<N_NODES, 128, 0, stream>>>(rp, acol, aval, sup2, b2, out_h, 0);

    // Head: y = h2 @ fc_w + fc_b -> out_y
    fc_kernel<<<(N_NODES * N_CLASS + 255) / 256, 256, 0, stream>>>(out_h, fcw, fcb, out_y);
}
